// MoEMLP_16999480557827
// MI455X (gfx1250) — matrix operand order repaired and checked by reading, not compiled, not run on hardware
//
#include <hip/hip_runtime.h>
#include <hip/hip_bf16.h>
#include <stdint.h>

// Problem sizes (fixed by the reference)
#define Dk 1024
#define Hk 4096
#define Ek 8
#define Tk 4096   // B*N = 2*2048
#define TM 128
#define TN 128
#define KC 64     // k-chunk per pipeline stage (2 wmma k-steps)

typedef __attribute__((ext_vector_type(16))) __bf16 v16bf;
typedef __attribute__((ext_vector_type(8)))  float  v8f;
typedef __attribute__((ext_vector_type(4)))  int    v4i;

struct __align__(16) U4 { unsigned int x, y, z, w; };
union Frag { U4 u[2]; v16bf v; };   // 32 bytes: one bf16 16x16x32 A/B fragment per lane

#define AS1 __attribute__((address_space(1)))
#define AS3 __attribute__((address_space(3)))

#if defined(__has_builtin)
#if __has_builtin(__builtin_amdgcn_global_load_async_to_lds_b128) && \
    __has_builtin(__builtin_amdgcn_s_wait_asynccnt)
#define USE_ASYNC_LDS 1
#endif
#endif

__device__ __forceinline__ unsigned short f2bf(float f) {
  unsigned int u = __float_as_uint(f);
  u += 0x7FFFu + ((u >> 16) & 1u);         // round-to-nearest-even
  return (unsigned short)(u >> 16);
}
__device__ __forceinline__ unsigned int pack2(float a, float b) {
  return (unsigned int)f2bf(a) | ((unsigned int)f2bf(b) << 16);
}
__device__ __forceinline__ float gelu_tanh(float v) {
  const float c = 0.7978845608028654f;     // sqrt(2/pi)
  float u = c * (v + 0.044715f * v * v * v);
  return 0.5f * v * (1.0f + tanhf(u));
}
__device__ __forceinline__ void store_out(unsigned short* p, float v) { *p = f2bf(v); }
__device__ __forceinline__ void store_out(float* p, float v) { *p = v; }

// ---------------------------------------------------------------- routing ---
__global__ void moe_zero_counts(int* __restrict__ counts) {
  if (threadIdx.x < Ek) counts[threadIdx.x] = 0;
}

// one wave32 per token; 8 waves per block
__global__ __launch_bounds__(256) void moe_router(
    const float* __restrict__ x, const float* __restrict__ Wg,
    const float* __restrict__ bg, int* __restrict__ counts,
    int* __restrict__ bucket) {
  const int wave = threadIdx.x >> 5;
  const int lane = threadIdx.x & 31;
  const int t = blockIdx.x * 8 + wave;

  float acc[Ek];
#pragma unroll
  for (int e = 0; e < Ek; ++e) acc[e] = 0.0f;

  const float* xr = x + (size_t)t * Dk;
  for (int d = lane * 4; d < Dk; d += 128) {
    float4 xv = *(const float4*)(xr + d);
    const float* f = (const float*)&xv;
#pragma unroll
    for (int j = 0; j < 4; ++j) {
      const float xs = f[j];
      const float* wgr = Wg + (size_t)(d + j) * Ek;
#pragma unroll
      for (int e = 0; e < Ek; ++e) acc[e] += xs * wgr[e];
    }
  }
#pragma unroll
  for (int off = 16; off > 0; off >>= 1)
#pragma unroll
    for (int e = 0; e < Ek; ++e) acc[e] += __shfl_down(acc[e], off, 32);

  if (lane == 0) {
    int best = 0;
    float bv = acc[0] + bg[0];
#pragma unroll
    for (int e = 1; e < Ek; ++e) {
      float v = acc[e] + bg[e];
      if (v > bv) { bv = v; best = e; }    // strict > : first max (jnp.argmax)
    }
    int pos = atomicAdd(&counts[best], 1);
    bucket[best * Tk + pos] = t;
  }
}

// ----------------------------------------------------- one-shot conversion ---
// x [T,D] f32 -> bf16 (elementwise, coalesced)
__global__ __launch_bounds__(256) void cvt_x(const float* __restrict__ x,
                                             unsigned short* __restrict__ xb) {
  const size_t i = ((size_t)blockIdx.x * 256 + threadIdx.x) * 8;
  float4 f0 = *(const float4*)(x + i);
  float4 f1 = *(const float4*)(x + i + 4);
  U4 q = { pack2(f0.x, f0.y), pack2(f0.z, f0.w), pack2(f1.x, f1.y), pack2(f1.z, f1.w) };
  *(U4*)(xb + i) = q;
}

// src [E][Rows][Cols] f32  ->  dst [E][Cols][Rows] bf16   (64x64 LDS tile)
__global__ __launch_bounds__(256) void convert_transpose(
    const float* __restrict__ src, unsigned short* __restrict__ dst,
    int Rows, int Cols) {
  __shared__ unsigned short tile[64][72];   // [col][row], padded vs bank conflicts
  const int e = blockIdx.z;
  const int r0 = blockIdx.x * 64;
  const int c0 = blockIdx.y * 64;

  const float* s = src + ((size_t)e * Rows + r0) * Cols + c0;
  const int lr = threadIdx.x >> 2;             // 0..63 local row
  const int lc = (threadIdx.x & 3) * 16;       // 16 cols per thread
  float4 g0 = ((const float4*)(s + (size_t)lr * Cols + lc))[0];
  float4 g1 = ((const float4*)(s + (size_t)lr * Cols + lc))[1];
  float4 g2 = ((const float4*)(s + (size_t)lr * Cols + lc))[2];
  float4 g3 = ((const float4*)(s + (size_t)lr * Cols + lc))[3];
  float gv[16] = {g0.x,g0.y,g0.z,g0.w, g1.x,g1.y,g1.z,g1.w,
                  g2.x,g2.y,g2.z,g2.w, g3.x,g3.y,g3.z,g3.w};
#pragma unroll
  for (int i = 0; i < 16; ++i) tile[lc + i][lr] = f2bf(gv[i]);
  __syncthreads();

  unsigned short* d = dst + ((size_t)e * Cols + c0) * Rows + r0;
  const int oc = threadIdx.x >> 2;             // output row (= source col)
  const int orr = (threadIdx.x & 3) * 16;      // 16 contiguous elements
  U4 q0 = *(const U4*)&tile[oc][orr];
  U4 q1 = *(const U4*)&tile[oc][orr + 8];
  *(U4*)(d + (size_t)oc * Rows + orr)     = q0;
  *(U4*)(d + (size_t)oc * Rows + orr + 8) = q1;
}

// -------------------------------------------------------- grouped GEMM ------
// A [T,KDIM] bf16 gathered by bucket;  Bt [E][NDIM][KDIM] bf16 (pre-transposed)
// 128x128 tile, 8 waves x (2x4) wmma C-tiles, double-buffered async LDS pipeline
template <int KDIM, int NDIM, bool GELU, typename OutT>
__global__ __launch_bounds__(256) void moe_gemm(
    const unsigned short* __restrict__ Ab, const unsigned short* __restrict__ Bt,
    const int* __restrict__ counts, const int* __restrict__ bucket,
    OutT* __restrict__ outp) {
  __shared__ unsigned short sA[2][TM * KC];   // [row][k] bf16, 16 KB per stage
  __shared__ unsigned short sB[2][TN * KC];   // [col][k] bf16, 16 KB per stage
  __shared__ int sTok[TM];

  const int e   = blockIdx.x >> 5;         // 32 row tiles per expert (worst case)
  const int m0  = (blockIdx.x & 31) * TM;
  const int cnt = counts[e];
  if (m0 >= cnt) return;
  const int n0  = blockIdx.y * TN;
  const int tid = threadIdx.x;

  if (tid < TM) {
    int idx = m0 + tid;
    sTok[tid] = bucket[e * Tk + (idx < cnt ? idx : m0)];  // clamp to valid row
  }
  __syncthreads();

  const int lane = tid & 31, wave = tid >> 5;
  const int wr = wave & 3;                 // rows  wr*32 .. +31 (2 C-tiles)
  const int wc = wave >> 2;                // cols  wc*64 .. +63 (4 C-tiles)

  v8f zero = {0.f, 0.f, 0.f, 0.f, 0.f, 0.f, 0.f, 0.f};
  v8f acc[2][4];
#pragma unroll
  for (int i = 0; i < 2; ++i)
#pragma unroll
    for (int j = 0; j < 4; ++j) acc[i][j] = zero;

  // staging: each thread moves 4x16B of A and of B per stage
  const int srow = tid >> 1, shalf = (tid & 1) * 32;
  const unsigned short* aSrc = Ab + (size_t)sTok[srow] * KDIM + shalf;
  const unsigned short* bSrc = Bt + ((size_t)e * NDIM + n0 + srow) * KDIM + shalf;
  const int sOff = srow * KC + shalf;

  auto stage = [&](int buf, int k0) {
#ifdef USE_ASYNC_LDS
#pragma unroll
    for (int c = 0; c < 4; ++c) {
      __builtin_amdgcn_global_load_async_to_lds_b128(
          (AS1 v4i*)(aSrc + k0 + c * 8), (AS3 v4i*)(&sA[buf][sOff + c * 8]), 0, 0);
      __builtin_amdgcn_global_load_async_to_lds_b128(
          (AS1 v4i*)(bSrc + k0 + c * 8), (AS3 v4i*)(&sB[buf][sOff + c * 8]), 0, 0);
    }
#else
#pragma unroll
    for (int c = 0; c < 4; ++c) {
      *(U4*)&sA[buf][sOff + c * 8] = *(const U4*)(aSrc + k0 + c * 8);
      *(U4*)&sB[buf][sOff + c * 8] = *(const U4*)(bSrc + k0 + c * 8);
    }
#endif
  };

  stage(0, 0);   // prologue: fill stage 0

  const int mrow = lane & 15;
  const int khA  = (lane >> 4) * 8;        // A: lanes 0-15 -> K 0-7/16-23, hi -> 8-15/24-31
  const int koB  = (lane >> 4) * 8;        // B: lanes 0-15 -> K 0-7/16-23, hi -> 8-15/24-31

  for (int k0 = 0; k0 < KDIM; k0 += KC) {
    const int cur = (k0 / KC) & 1;
#ifdef USE_ASYNC_LDS
    __builtin_amdgcn_s_wait_asynccnt(0);   // my writes into cur have landed
#endif
    __syncthreads();                       // everyone's writes landed, prior reads done
    if (k0 + KC < KDIM) stage(cur ^ 1, k0 + KC);  // DMA next chunk during compute

#pragma unroll
    for (int ks = 0; ks < KC; ks += 32) {
      Frag a[2], b[4];
#pragma unroll
      for (int rt = 0; rt < 2; ++rt) {
        int row = wr * 32 + rt * 16 + mrow;
        a[rt].u[0] = *(const U4*)&sA[cur][row * KC + ks + khA];
        a[rt].u[1] = *(const U4*)&sA[cur][row * KC + ks + khA + 16];
      }
#pragma unroll
      for (int ct = 0; ct < 4; ++ct) {
        int col = wc * 64 + ct * 16 + mrow;
        b[ct].u[0] = *(const U4*)&sB[cur][col * KC + ks + koB];
        b[ct].u[1] = *(const U4*)&sB[cur][col * KC + ks + koB + 16];
      }
#pragma unroll
      for (int rt = 0; rt < 2; ++rt)
#pragma unroll
        for (int ct = 0; ct < 4; ++ct)
          acc[rt][ct] = __builtin_amdgcn_wmma_f32_16x16x32_bf16(
              false, a[rt].v, false, b[ct].v, (short)0, acc[rt][ct], false, false);
    }
  }

  // epilogue: C layout vgpr r -> M = r (lanes 0-15) / r+8 (lanes 16-31), N = lane&15
  const int nlow = lane & 15;
  const int mhi  = (lane >> 4) * 8;
#pragma unroll
  for (int rt = 0; rt < 2; ++rt)
#pragma unroll
    for (int ct = 0; ct < 4; ++ct)
#pragma unroll
      for (int r = 0; r < 8; ++r) {
        int m = wr * 32 + rt * 16 + r + mhi;
        if (m0 + m < cnt) {
          int n = n0 + wc * 64 + ct * 16 + nlow;
          float v = acc[rt][ct][r];
          if (GELU) v = gelu_tanh(v);
          store_out(outp + (size_t)sTok[m] * NDIM + n, v);
        }
      }
}

// ---------------------------------------------------------------- launch ---
extern "C" void kernel_launch(void* const* d_in, const int* in_sizes, int n_in,
                              void* d_out, int out_size, void* d_ws, size_t ws_size,
                              hipStream_t stream) {
  (void)in_sizes; (void)n_in; (void)out_size; (void)ws_size;
  const float* x  = (const float*)d_in[0];   // [T, D]
  const float* W1 = (const float*)d_in[1];   // [E, D, H]
  const float* W2 = (const float*)d_in[2];   // [E, H, D]
  const float* Wg = (const float*)d_in[3];   // [D, E]
  const float* bg = (const float*)d_in[4];   // [E]
  float* out = (float*)d_out;                // [T, D] f32

  char* ws = (char*)d_ws;
  int* counts = (int*)ws;                                        // 8 ints
  int* bucket = (int*)(ws + 256);                                // E*T ints (128 KB)
  size_t off = 256 + (size_t)Ek * Tk * 4;
  unsigned short* xb   = (unsigned short*)(ws + off); off += (size_t)Tk * Dk * 2;   // 8 MB
  unsigned short* hbuf = (unsigned short*)(ws + off); off += (size_t)Tk * Hk * 2;   // 32 MB
  unsigned short* W1T  = (unsigned short*)(ws + off); off += (size_t)Ek * Dk * Hk * 2; // 64 MB
  unsigned short* W2T  = (unsigned short*)(ws + off);            // 64 MB

  moe_zero_counts<<<1, 64, 0, stream>>>(counts);
  moe_router<<<Tk / 8, 256, 0, stream>>>(x, Wg, bg, counts, bucket);
  cvt_x<<<(Tk * Dk) / 2048, 256, 0, stream>>>(x, xb);
  // W1 [E][D][H] -> W1T [E][H][D] ; W2 [E][H][D] -> W2T [E][D][H]
  convert_transpose<<<dim3(Dk / 64, Hk / 64, Ek), 256, 0, stream>>>(W1, W1T, Dk, Hk);
  convert_transpose<<<dim3(Hk / 64, Dk / 64, Ek), 256, 0, stream>>>(W2, W2T, Hk, Dk);

  // h = gelu(x @ W1[e])  : K=D, N=H, bf16 out
  moe_gemm<Dk, Hk, true, unsigned short>
      <<<dim3(Ek * (Tk / TM), Hk / TN), 256, 0, stream>>>(xb, W1T, counts, bucket, hbuf);
  // out = h @ W2[e]      : K=H, N=D, f32 out
  moe_gemm<Hk, Dk, false, float>
      <<<dim3(Ek * (Tk / TM), Dk / TN), 256, 0, stream>>>(hbuf, W2T, counts, bucket, out);
}
